// BERT_86809878987310
// MI455X (gfx1250) — matrix operand order repaired and checked by reading, not compiled, not run on hardware
//
#include <hip/hip_runtime.h>
#include <math.h>

// ---------------- model constants ----------------
#define DD     768
#define HH     12
#define DKV    64
#define SS     512
#define BBATCH 16
#define DFFN   3072
#define NLAYER 12
#define NPREDN 20
#define VVOC   32000

// ---------------- WMMA types ----------------
typedef __attribute__((ext_vector_type(16))) __bf16       v16bf;
typedef __attribute__((ext_vector_type(8)))  float        v8f;
typedef __attribute__((ext_vector_type(4)))  unsigned int u32x4;
typedef __attribute__((ext_vector_type(2)))  unsigned int u32x2;

union Frag16 { v16bf v; u32x4 q[2]; unsigned int u[8]; };

__device__ __forceinline__ unsigned short f2bf(float f) {
  union { float f; unsigned int u; } x; x.f = f;
  unsigned int r = x.u + 0x7FFFu + ((x.u >> 16) & 1u); // round-to-nearest-even
  return (unsigned short)(r >> 16);
}
__device__ __forceinline__ unsigned int pk2bf(float a, float b) {
  return (unsigned int)f2bf(a) | ((unsigned int)f2bf(b) << 16);
}
__device__ __forceinline__ float bf2f(unsigned short h) {
  union { unsigned int u; float f; } x; x.u = ((unsigned int)h) << 16; return x.f;
}
__device__ __forceinline__ v8f wmma_bf(const Frag16& a, const Frag16& b, v8f c) {
  return __builtin_amdgcn_wmma_f32_16x16x32_bf16(false, a.v, false, b.v,
                                                 (short)0, c, false, false);
}
__device__ __forceinline__ v8f v8zero() {
  v8f z; for (int i = 0; i < 8; i++) z[i] = 0.f; return z;
}

// ---- CDNA5 async global->LDS copy (16B per lane), tracked by ASYNCcnt ----
__device__ __forceinline__ void async_cp16(void* lds, const void* gsrc) {
  unsigned int      ldsoff = (unsigned int)(uintptr_t)lds;     // low 32b = LDS byte addr
  unsigned long long gaddr = (unsigned long long)(uintptr_t)gsrc;
  asm volatile("global_load_async_to_lds_b128 %0, %1, off"
               :: "v"(ldsoff), "v"(gaddr) : "memory");
}
__device__ __forceinline__ void async_wait0() {
  asm volatile("s_wait_asynccnt 0x0" ::: "memory");
}

// ======================================================================
// GEMM: C[M,N] = act(A_bf16[M,K](row stride lda) @ W_f32 + bias + resid)
//   W: bT==0 -> [K,N] row-major f32 ; bT==1 -> [N,K] row-major f32
//   act: 0=none 1=gelu(exact) 2=tanh ; outf32: 1 -> f32 C, else bf16 C
// Block 128 threads (4 waves); 64x64 tile, BK=32; wave w -> rows [16w,16w+16).
// Full tiles take an unguarded vector path with async A staging.
// ======================================================================
#define LAS 40  // LDS row stride (shorts) = 32 + 8 pad -> 80B (16B aligned)

__global__ __launch_bounds__(128)
void gemm_wmma_bf16(const unsigned short* __restrict__ A, int lda,
                    const float* __restrict__ W,
                    const float* __restrict__ bias,
                    const unsigned short* __restrict__ resid,
                    void* __restrict__ Cout,
                    int M, int N, int K, int act, int bT, int outf32)
{
  __shared__ __align__(16) unsigned short Asm[64 * LAS];
  __shared__ __align__(16) unsigned short Bsm[64 * LAS];   // BT layout: [n][k]

  const int tid  = threadIdx.x;
  const int lane = tid & 31;
  const int wave = tid >> 5;
  const int lr   = lane & 15;
  const int lhi  = lane >> 4;
  const int row0 = blockIdx.y * 64;
  const int col0 = blockIdx.x * 64;
  const bool full = (row0 + 64 <= M) && (col0 + 64 <= N);

  v8f acc[4];
  for (int i = 0; i < 4; i++) acc[i] = v8zero();

  for (int kk = 0; kk < K; kk += 32) {
    __syncthreads();
    if (full) {
      // ---- A tile: pure byte copy, async global->LDS (64 rows x 64B) ----
#pragma unroll
      for (int p = tid; p < 256; p += 128) {
        int m = p >> 2, c = p & 3;                       // 4 x 16B chunks per row
        async_cp16(Asm + m * LAS + 8 * c,
                   A + (size_t)(row0 + m) * lda + kk + 8 * c);
      }
      // ---- B tile: f32 -> bf16 convert into BT[n][k] ----
      if (bT) {
#pragma unroll
        for (int p = tid; p < 512; p += 128) {
          int n = p >> 3, c = p & 7;                     // 4 k per chunk
          float4 r = *(const float4*)(W + (size_t)(col0 + n) * K + kk + 4 * c);
          u32x2 d; d[0] = pk2bf(r.x, r.y); d[1] = pk2bf(r.z, r.w);
          *(u32x2*)(Bsm + n * LAS + 4 * c) = d;
        }
      } else {
#pragma unroll
        for (int p = tid; p < 256; p += 128) {
          int kp = p >> 4;                               // k-pair 0..15
          int n  = (p & 15) * 4;                         // 4 consecutive n
          const float* w0 = W + (size_t)(kk + 2 * kp) * N + col0 + n;
          float4 r0 = *(const float4*)w0;
          float4 r1 = *(const float4*)(w0 + N);
          ((unsigned int*)(Bsm + (n + 0) * LAS))[kp] = pk2bf(r0.x, r1.x);
          ((unsigned int*)(Bsm + (n + 1) * LAS))[kp] = pk2bf(r0.y, r1.y);
          ((unsigned int*)(Bsm + (n + 2) * LAS))[kp] = pk2bf(r0.z, r1.z);
          ((unsigned int*)(Bsm + (n + 3) * LAS))[kp] = pk2bf(r0.w, r1.w);
        }
      }
      async_wait0();
    } else {
      // ---- guarded path (edge tiles: pooler/cls heads) ----
      for (int p = tid; p < 512; p += 128) {
        int m = p >> 3, c = p & 7;
        int gm = row0 + m;
        u32x2 d; d[0] = 0u; d[1] = 0u;
        if (gm < M) d = *(const u32x2*)(A + (size_t)gm * lda + kk + 4 * c);
        *(u32x2*)(Asm + m * LAS + 4 * c) = d;
      }
      for (int p = tid; p < 1024; p += 128) {
        int n = p >> 4, kp = p & 15;
        int gn = col0 + n;
        float b0 = 0.f, b1 = 0.f;
        if (gn < N) {
          if (bT) {
            const float* bp = W + (size_t)gn * K + kk + 2 * kp;
            b0 = bp[0]; b1 = bp[1];
          } else {
            const float* bp = W + (size_t)(kk + 2 * kp) * N + gn;
            b0 = bp[0]; b1 = bp[N];
          }
        }
        ((unsigned int*)(Bsm + n * LAS))[kp] = pk2bf(b0, b1);
      }
    }
    __syncthreads();

    // ---- compute ----
    const unsigned short* Arow = Asm + (wave * 16 + lr) * LAS;
    Frag16 af;
    af.q[0] = *(const u32x4*)(Arow + lhi * 8);        // K = lhi*8 + 0..7
    af.q[1] = *(const u32x4*)(Arow + 16 + lhi * 8);   // K = 16 + lhi*8 + 0..7
#pragma unroll
    for (int nt = 0; nt < 4; nt++) {
      const unsigned short* Brow = Bsm + (nt * 16 + lr) * LAS;
      Frag16 bf;
      bf.q[0] = *(const u32x4*)(Brow + lhi * 8);
      bf.q[1] = *(const u32x4*)(Brow + 16 + lhi * 8);
      acc[nt] = wmma_bf(af, bf, acc[nt]);
    }
  }

  // ---- epilogue: bias + residual + activation ----
  float*          Cf = (float*)Cout;
  unsigned short* Ch = (unsigned short*)Cout;
#pragma unroll
  for (int nt = 0; nt < 4; nt++) {
    int n = col0 + nt * 16 + lr;
    if (n >= N) continue;
    float bv = bias ? bias[n] : 0.f;
#pragma unroll
    for (int r = 0; r < 8; r++) {
      int m = row0 + wave * 16 + lhi * 8 + r;
      if (m >= M) continue;
      float v = acc[nt][r] + bv;
      if (resid) v += bf2f(resid[(size_t)m * N + n]);
      if (act == 1)      v = 0.5f * v * (1.f + erff(v * 0.70710678118654752f));
      else if (act == 2) v = tanhf(v);
      if (outf32) Cf[(size_t)m * N + n] = v;
      else        Ch[(size_t)m * N + n] = f2bf(v);
    }
  }
}

// ======================================================================
// Attention (bf16 Q/K/V -> bf16 ctx): per (b,h,64-query tile).
// Two-pass softmax in big WGP LDS. 128 threads, wave w -> rows [16w,16w+16).
// ======================================================================
#define QSTR 72    // Q LDS row stride (shorts), 144B
#define SSTR 516   // score LDS row stride (floats)
#define PSTR 520   // P / V^T LDS row stride (shorts), 1040B
#define ATTN_SMEM (64*QSTR*2 + 64*SSTR*4 + 64*PSTR*2 + 64*PSTR*2 + 512*4) // 276480B

__global__ __launch_bounds__(128)
void attn_wmma_bf16(const unsigned short* __restrict__ Q,
                    const unsigned short* __restrict__ Kx,
                    const unsigned short* __restrict__ V,
                    const int* __restrict__ ids,
                    unsigned short* __restrict__ O)
{
  extern __shared__ __align__(16) unsigned char smem[];
  unsigned short* Qs = (unsigned short*)smem;                         // 64 x QSTR
  float*          Sc = (float*)(smem + 64 * QSTR * 2);                // 64 x SSTR
  unsigned short* Pb = (unsigned short*)((unsigned char*)Sc + 64 * SSTR * 4);
  unsigned short* Vt = Pb + 64 * PSTR;                                // 64 dv x keys
  int*            mk = (int*)(Vt + 64 * PSTR);                        // 512 pad flags

  const int tid  = threadIdx.x;
  const int lane = tid & 31;
  const int wave = tid >> 5;
  const int lr   = lane & 15;
  const int lhi  = lane >> 4;
  const int q0   = blockIdx.x * 64;
  const int b    = blockIdx.y / HH;
  const int h    = blockIdx.y % HH;
  const size_t base = (size_t)b * SS * DD + (size_t)h * DKV;

  // ---- Q tile: async byte copy (64q x 128B) ----
#pragma unroll
  for (int p = tid; p < 512; p += 128) {
    int m = p >> 3, c = p & 7;
    async_cp16(Qs + m * QSTR + 8 * c,
               Q + base + (size_t)(q0 + m) * DD + 8 * c);
  }
  // ---- V transposed into Vt[dv][key]: b128 read + 8 b16 LDS scatters ----
  for (int p = tid; p < 4096; p += 128) {
    int key = p >> 3, c = p & 7;
    union { u32x4 q; unsigned short s[8]; } r;
    r.q = *(const u32x4*)(V + base + (size_t)key * DD + 8 * c);
#pragma unroll
    for (int i = 0; i < 8; i++) Vt[(8 * c + i) * PSTR + key] = r.s[i];
  }
  for (int i = tid; i < SS; i += 128) mk[i] = (ids[b * SS + i] == 0);
  async_wait0();
  __syncthreads();

  // ---- phase 1: scores = (Q K^T) * scale, pad-masked -> LDS ----
  const int m0 = wave * 16;
  const unsigned short* Arow = Qs + (m0 + lr) * QSTR;
  Frag16 a0, a1;
  a0.q[0] = *(const u32x4*)(Arow + lhi * 8);
  a0.q[1] = *(const u32x4*)(Arow + 16 + lhi * 8);
  a1.q[0] = *(const u32x4*)(Arow + 32 + lhi * 8);
  a1.q[1] = *(const u32x4*)(Arow + 48 + lhi * 8);

  for (int nt = 0; nt < 32; nt++) {
    int key = nt * 16 + lr;
    // B = K^T: bf16 K rows are already [key][dk] -> K = lhi*8 + 0..7, 16 + lhi*8 + 0..7
    const unsigned short* kb = Kx + base + (size_t)key * DD;
    Frag16 b0, b1;
    b0.q[0] = *(const u32x4*)(kb + lhi * 8);
    b0.q[1] = *(const u32x4*)(kb + 16 + lhi * 8);
    b1.q[0] = *(const u32x4*)(kb + 32 + lhi * 8);
    b1.q[1] = *(const u32x4*)(kb + 48 + lhi * 8);
    v8f c = v8zero();
    c = wmma_bf(a0, b0, c);
    c = wmma_bf(a1, b1, c);
    int masked = mk[key];
#pragma unroll
    for (int r = 0; r < 8; r++) {
      int m = m0 + lhi * 8 + r;
      Sc[m * SSTR + key] = masked ? -__builtin_inff() : c[r] * 0.125f;
    }
  }
  __syncthreads();

  // ---- phase 2: row softmax; requantize P to bf16 ----
  if (tid < 64) {
    float* srow = Sc + tid * SSTR;
    float mx = -__builtin_inff();
    for (int j = 0; j < SS; j++) mx = fmaxf(mx, srow[j]);
    float sum = 0.f;
    for (int j = 0; j < SS; j++) { float e = __expf(srow[j] - mx); srow[j] = e; sum += e; }
    float inv = 1.f / sum;
    unsigned int* prow = (unsigned int*)(Pb + tid * PSTR);
    for (int j = 0; j < SS / 2; j++)
      prow[j] = pk2bf(srow[2 * j] * inv, srow[2 * j + 1] * inv);
  }
  __syncthreads();

  // ---- phase 3: ctx = P @ V ----
  v8f acc[4];
  for (int i = 0; i < 4; i++) acc[i] = v8zero();
  const unsigned short* Prow = Pb + (m0 + lr) * PSTR;
  for (int ks = 0; ks < 16; ks++) {
    Frag16 af;
    af.q[0] = *(const u32x4*)(Prow + ks * 32 + lhi * 8);
    af.q[1] = *(const u32x4*)(Prow + ks * 32 + 16 + lhi * 8);
#pragma unroll
    for (int nt = 0; nt < 4; nt++) {
      const unsigned short* Vrow = Vt + (nt * 16 + lr) * PSTR + ks * 32;
      Frag16 bf;
      bf.q[0] = *(const u32x4*)(Vrow + lhi * 8);
      bf.q[1] = *(const u32x4*)(Vrow + 16 + lhi * 8);
      acc[nt] = wmma_bf(af, bf, acc[nt]);
    }
  }
#pragma unroll
  for (int nt = 0; nt < 4; nt++) {
    int dv = nt * 16 + lr;
#pragma unroll
    for (int r = 0; r < 8; r++) {
      int m = m0 + lhi * 8 + r;
      O[base + (size_t)(q0 + m) * DD + dv] = f2bf(acc[nt][r]);
    }
  }
}

// ======================================================================
// Fused embedding gather + LayerNorm -> bf16 X. One block per token row.
// ======================================================================
__global__ __launch_bounds__(256)
void embed_ln(const int* __restrict__ ids, const int* __restrict__ segs,
              const float* __restrict__ tok, const float* __restrict__ pos,
              const float* __restrict__ seg, const float* __restrict__ g,
              const float* __restrict__ bb, unsigned short* __restrict__ X)
{
  __shared__ float red[256];
  const int row = blockIdx.x;
  const int s   = row & (SS - 1);
  const int id  = ids[row], sg = segs[row];
  const float* tp = tok + (size_t)id * DD;
  const float* pp = pos + (size_t)s  * DD;
  const float* sp = seg + (size_t)sg * DD;
  float v[3]; float loc = 0.f;
#pragma unroll
  for (int i = 0; i < 3; i++) {
    int d = threadIdx.x + i * 256;
    float x = tp[d] + pp[d] + sp[d];
    v[i] = x; loc += x;
  }
  red[threadIdx.x] = loc; __syncthreads();
  for (int o = 128; o > 0; o >>= 1) { if (threadIdx.x < o) red[threadIdx.x] += red[threadIdx.x + o]; __syncthreads(); }
  float mean = red[0] * (1.f / DD);
  __syncthreads();
  loc = 0.f;
#pragma unroll
  for (int i = 0; i < 3; i++) { float d = v[i] - mean; loc += d * d; }
  red[threadIdx.x] = loc; __syncthreads();
  for (int o = 128; o > 0; o >>= 1) { if (threadIdx.x < o) red[threadIdx.x] += red[threadIdx.x + o]; __syncthreads(); }
  float inv = rsqrtf(red[0] * (1.f / DD) + 1e-5f);
  unsigned short* xr = X + (size_t)row * DD;
#pragma unroll
  for (int i = 0; i < 3; i++) {
    int d = threadIdx.x + i * 256;
    xr[d] = f2bf(g[d] * (v[i] - mean) * inv + bb[d]);
  }
}

// LayerNorm over bf16 rows of [M, 768] -> bf16
__global__ __launch_bounds__(256)
void ln_rows(const unsigned short* __restrict__ in, const float* __restrict__ g,
             const float* __restrict__ bb, unsigned short* __restrict__ out)
{
  __shared__ float red[256];
  const int row = blockIdx.x;
  const unsigned short* ir = in + (size_t)row * DD;
  float v[3]; float loc = 0.f;
#pragma unroll
  for (int i = 0; i < 3; i++) { int d = threadIdx.x + i * 256; float x = bf2f(ir[d]); v[i] = x; loc += x; }
  red[threadIdx.x] = loc; __syncthreads();
  for (int o = 128; o > 0; o >>= 1) { if (threadIdx.x < o) red[threadIdx.x] += red[threadIdx.x + o]; __syncthreads(); }
  float mean = red[0] * (1.f / DD);
  __syncthreads();
  loc = 0.f;
#pragma unroll
  for (int i = 0; i < 3; i++) { float d = v[i] - mean; loc += d * d; }
  red[threadIdx.x] = loc; __syncthreads();
  for (int o = 128; o > 0; o >>= 1) { if (threadIdx.x < o) red[threadIdx.x] += red[threadIdx.x + o]; __syncthreads(); }
  float inv = rsqrtf(red[0] * (1.f / DD) + 1e-5f);
  unsigned short* orow = out + (size_t)row * DD;
#pragma unroll
  for (int i = 0; i < 3; i++) { int d = threadIdx.x + i * 256; orow[d] = f2bf(g[d] * (v[i] - mean) * inv + bb[d]); }
}

// gather rows x[b, masked_pos[b,i]] -> out[b*NPRED+i]  (bf16)
__global__ __launch_bounds__(256)
void gather_masked(const unsigned short* __restrict__ X, const int* __restrict__ mpos,
                   unsigned short* __restrict__ out)
{
  const int j = blockIdx.x;
  const int b = j / NPREDN;
  const int p = mpos[j];
  const unsigned short* src = X + ((size_t)b * SS + p) * DD;
  unsigned short* dst = out + (size_t)j * DD;
  for (int d = threadIdx.x; d < DD; d += 256) dst[d] = src[d];
}

// ---------------- host-side launcher ----------------
static inline void gemm(hipStream_t st, const unsigned short* A, int lda,
                        const float* W, const float* bias,
                        const unsigned short* resid, void* C,
                        int M, int N, int K, int act, int bT, int outf32)
{
  dim3 grid((N + 63) / 64, (M + 63) / 64);
  gemm_wmma_bf16<<<grid, 128, 0, st>>>(A, lda, W, bias, resid, C, M, N, K, act, bT, outf32);
}

extern "C" void kernel_launch(void* const* d_in, const int* in_sizes, int n_in,
                              void* d_out, int out_size, void* d_ws, size_t ws_size,
                              hipStream_t stream)
{
  (void)in_sizes; (void)n_in; (void)out_size; (void)ws_size;
  const int*   input_ids   = (const int*)d_in[0];
  const int*   segment_ids = (const int*)d_in[1];
  const int*   masked_pos  = (const int*)d_in[2];
  const float* tok   = (const float*)d_in[3];
  const float* pos   = (const float*)d_in[4];
  const float* seg   = (const float*)d_in[5];
  const float* emb_g = (const float*)d_in[6];
  const float* emb_b = (const float*)d_in[7];
  const float* Wq = (const float*)d_in[8];   const float* bq = (const float*)d_in[9];
  const float* Wk = (const float*)d_in[10];  const float* bk = (const float*)d_in[11];
  const float* Wv = (const float*)d_in[12];  const float* bv = (const float*)d_in[13];
  const float* Wo = (const float*)d_in[14];  const float* bo = (const float*)d_in[15];
  const float* ln1g = (const float*)d_in[16]; const float* ln1b = (const float*)d_in[17];
  const float* W1 = (const float*)d_in[18];  const float* b1 = (const float*)d_in[19];
  const float* W2 = (const float*)d_in[20];  const float* b2 = (const float*)d_in[21];
  const float* ln2g = (const float*)d_in[22]; const float* ln2b = (const float*)d_in[23];
  const float* pool_W = (const float*)d_in[24]; const float* pool_b = (const float*)d_in[25];
  const float* cls_W  = (const float*)d_in[26]; const float* cls_b  = (const float*)d_in[27];
  const float* lin_W  = (const float*)d_in[28]; const float* lin_b  = (const float*)d_in[29];

  const int T = BBATCH * SS;                 // 8192 token rows
  unsigned short* ws = (unsigned short*)d_ws;
  unsigned short* X   = ws;                           // [T, D] bf16
  unsigned short* Qb  = X   + (size_t)T * DD;
  unsigned short* Kb  = Qb  + (size_t)T * DD;
  unsigned short* Vb  = Kb  + (size_t)T * DD;
  unsigned short* Cb  = Vb  + (size_t)T * DD;         // attention context
  unsigned short* FF  = Cb  + (size_t)T * DD;         // [T, DFF]
  unsigned short* HM  = FF  + (size_t)T * DFFN;       // [320, D]
  unsigned short* HM2 = HM  + (size_t)BBATCH * NPREDN * DD;
  unsigned short* PO  = HM2 + (size_t)BBATCH * NPREDN * DD;  // pooled [16, D]
  unsigned short* Tb  = Qb;                           // reuse q as post-attn temp
  unsigned short* X2  = Kb;                           // reuse k as post-ln1 hidden

  (void)hipFuncSetAttribute((const void*)attn_wmma_bf16,
                            hipFuncAttributeMaxDynamicSharedMemorySize, ATTN_SMEM);

  embed_ln<<<T, 256, 0, stream>>>(input_ids, segment_ids, tok, pos, seg, emb_g, emb_b, X);

  for (int l = 0; l < NLAYER; l++) {
    const size_t wo  = (size_t)l * DD * DD;
    const size_t w1o = (size_t)l * DD * DFFN;
    const size_t w2o = (size_t)l * DFFN * DD;
    gemm(stream, X, DD, Wq + wo, bq + (size_t)l * DD, nullptr, Qb, T, DD, DD, 0, 0, 0);
    gemm(stream, X, DD, Wk + wo, bk + (size_t)l * DD, nullptr, Kb, T, DD, DD, 0, 0, 0);
    gemm(stream, X, DD, Wv + wo, bv + (size_t)l * DD, nullptr, Vb, T, DD, DD, 0, 0, 0);
    attn_wmma_bf16<<<dim3(SS / 64, BBATCH * HH), 128, ATTN_SMEM, stream>>>(
        Qb, Kb, Vb, input_ids, Cb);
    gemm(stream, Cb, DD, Wo + wo, bo + (size_t)l * DD, X, Tb, T, DD, DD, 0, 0, 0);
    ln_rows<<<T, 256, 0, stream>>>(Tb, ln1g + (size_t)l * DD, ln1b + (size_t)l * DD, X2);
    gemm(stream, X2, DD, W1 + w1o, b1 + (size_t)l * DFFN, nullptr, FF, T, DFFN, DD, 1, 0, 0);
    gemm(stream, FF, DFFN, W2 + w2o, b2 + (size_t)l * DD, X2, Tb, T, DD, DFFN, 0, 0, 0);
    ln_rows<<<T, 256, 0, stream>>>(Tb, ln2g + (size_t)l * DD, ln2b + (size_t)l * DD, X);
  }

  // ---- heads ----
  float* out_lm  = (float*)d_out;                                   // [16,20,32000] f32
  float* out_cls = out_lm + (size_t)BBATCH * NPREDN * VVOC;         // [16,2] f32

  // pooler: tanh(x[:,0] @ pool_W + pool_b)  (lda strides over S*D to pick s==0)
  gemm(stream, X, SS * DD, pool_W, pool_b, nullptr, PO, BBATCH, DD, DD, 2, 0, 0);
  gemm(stream, PO, DD, cls_W, cls_b, nullptr, out_cls, BBATCH, 2, DD, 0, 0, 1);

  // MLM head: gather -> gelu linear -> tied vocab projection (tok [V,D] = B^T)
  gather_masked<<<BBATCH * NPREDN, 256, 0, stream>>>(X, masked_pos, HM);
  gemm(stream, HM, DD, lin_W, lin_b, nullptr, HM2, BBATCH * NPREDN, DD, DD, 1, 0, 0);
  gemm(stream, HM2, DD, tok, nullptr, nullptr, out_lm, BBATCH * NPREDN, VVOC, DD, 0, 1, 1);
}
